// MaskedSelfAttention_52733608461094
// MI455X (gfx1250) — matrix operand order repaired and checked by reading, not compiled, not run on hardware
//
#include <hip/hip_runtime.h>
#include <stdint.h>

// Problem constants (match reference).
#define Bb 4
#define Cc 256
#define Nn 4096        // H*W = 64*64
#define Dd 32          // C/8

typedef __bf16 bf16;
typedef __attribute__((ext_vector_type(16))) bf16 v16bf;
typedef __attribute__((ext_vector_type(8)))  float v8f;

union FragB { v16bf v; uint4 q[2]; };   // 16 bf16 = 32 bytes = 2 x b128

__device__ __forceinline__ unsigned short f2bf(float f) {
    unsigned u = __float_as_uint(f);
    u = (u + 0x7FFFu + ((u >> 16) & 1u)) >> 16;   // round-to-nearest-even
    return (unsigned short)u;
}

// ---- DPP-based 16-lane (row) reductions: pure VALU, no LDS round-trips ----
// xor1 = quad_perm[1,0,3,2]=0xB1, xor2 = quad_perm[2,3,0,1]=0x4E,
// xor7 = ROW_HALF_MIRROR=0x141, xor15 = ROW_MIRROR=0x140.
// {1,2,7,15} span the 16-lane group -> full butterfly reduction in 4 steps.
template <int CTRL>
__device__ __forceinline__ float dpp_swap(float v) {
    return __uint_as_float(__builtin_amdgcn_update_dpp(
        __float_as_uint(v), __float_as_uint(v), CTRL, 0xF, 0xF, false));
}
__device__ __forceinline__ float rowmax16(float v) {
    v = fmaxf(v, dpp_swap<0xB1>(v));
    v = fmaxf(v, dpp_swap<0x4E>(v));
    v = fmaxf(v, dpp_swap<0x141>(v));
    v = fmaxf(v, dpp_swap<0x140>(v));
    return v;
}
__device__ __forceinline__ float rowsum16(float v) {
    v += dpp_swap<0xB1>(v);
    v += dpp_swap<0x4E>(v);
    v += dpp_swap<0x141>(v);
    v += dpp_swap<0x140>(v);
    return v;
}

// ---------------------------------------------------------------------------
// Scalar projections for Q and K (tiny: 0.13 G MACs each).
// Q stored row-major (B,N,D) bf16 with 1/sqrt(D) folded in (A-fragment friendly).
// K stored pre-packed in WMMA B-fragment layout:  KB[b][kt(N/16)][lane][e],
//   d = (lane/16)*16 + e, key = kt*16 + lane%16.
// ---------------------------------------------------------------------------
__global__ void proj_q_kernel(const float* __restrict__ x, const float* __restrict__ Wq,
                              const float* __restrict__ bq, unsigned short* __restrict__ Q) {
    int idx = blockIdx.x * 256 + threadIdx.x;        // B*N*D = 524288
    int n = idx & (Nn - 1);
    int d = (idx >> 12) & (Dd - 1);
    int b = idx >> 17;
    const float* xb = x + (size_t)b * Cc * Nn;
    float s = bq[d];
    #pragma unroll 8
    for (int c = 0; c < Cc; ++c) s += Wq[d * Cc + c] * xb[c * Nn + n];
    Q[((size_t)(b * Nn + n)) * Dd + d] = f2bf(s * 0.17677669529663687f);
}

__global__ void proj_k_kernel(const float* __restrict__ x, const float* __restrict__ Wk,
                              const float* __restrict__ bk, unsigned short* __restrict__ KB) {
    int idx = blockIdx.x * 256 + threadIdx.x;        // B*(N/16)*32*16 = 524288
    int e    = idx & 15;
    int lane = (idx >> 4) & 31;
    int kt   = (idx >> 9) & 255;
    int b    = idx >> 17;
    int d    = (lane >> 4) * 16 + e;
    int key  = kt * 16 + (lane & 15);
    const float* xb = x + (size_t)b * Cc * Nn;
    float s = bk[d];
    #pragma unroll 8
    for (int c = 0; c < Cc; ++c) s += Wk[d * Cc + c] * xb[c * Nn + key];
    KB[idx] = f2bf(s);
}

// ---------------------------------------------------------------------------
// WMMA path for V = Wv * X (1.07 G MACs -> matrix core).
// Step 1: pack Wv into A-fragment layout WvP[ct(16)][chunk(8)][lane][e] bf16:
//   c = ct*16 + lane%16, cc = chunk*32 + (lane/16)*16 + e.
// Step 2: pack x into B-fragment layout XB[b][chunk(8)][kt(N/16)][lane][e] bf16:
//   cc = chunk*32 + (lane/16)*16 + e, key = kt*16 + lane%16. (coalesced f32 reads)
// Step 3: WMMA GEMM; result written directly in attn's packed V-frag layout:
//   VB[b][kt2(N/32)][ct(C/16)][lane][e]: c = ct*16+lane%16, key = kt2*32+(lane/16)*16+e.
// ---------------------------------------------------------------------------
__global__ void pack_wv_kernel(const float* __restrict__ Wv, unsigned short* __restrict__ WvP) {
    int idx  = blockIdx.x * 256 + threadIdx.x;       // 16*8*32*16 = 65536
    int e    = idx & 15;
    int lane = (idx >> 4) & 31;
    int ch   = (idx >> 9) & 7;
    int ct   = idx >> 12;
    int c    = ct * 16 + (lane & 15);
    int cc   = ch * 32 + (lane >> 4) * 16 + e;
    WvP[idx] = f2bf(Wv[c * Cc + cc]);
}

__global__ void pack_x_kernel(const float* __restrict__ x, unsigned short* __restrict__ XB) {
    size_t idx = (size_t)blockIdx.x * 256 + threadIdx.x;   // B*C*N, key fastest (coalesced reads)
    int key = idx & (Nn - 1);
    int cc  = (idx >> 12) & (Cc - 1);
    int b   = idx >> 20;
    int ch = cc >> 5, w = cc & 31;
    int lane = (w >> 4) * 16 + (key & 15);
    int e    = w & 15;
    int kt   = key >> 4;
    XB[((((size_t)b * 8 + ch) * 256 + kt) * 32 + lane) * 16 + e] = f2bf(x[idx]);
}

__global__ __launch_bounds__(256)
void proj_v_wmma_kernel(const unsigned short* __restrict__ WvP,
                        const unsigned short* __restrict__ XB,
                        const float* __restrict__ bv,
                        unsigned short* __restrict__ VB) {
    __shared__ unsigned short ldsT[8][16 * 32];      // per-wave 16c x 32key bf16 tile

    const int lane = threadIdx.x & 31;
    const int wv   = threadIdx.x >> 5;
    const int gw   = blockIdx.x * 8 + wv;            // 4*16*128 = 8192 waves
    const int kt2  = gw & 127;
    const int ct   = (gw >> 7) & 15;
    const int b    = gw >> 11;
    const int lm = lane & 15, lh = lane >> 4;

    v8f acc0 = (v8f){0.f,0.f,0.f,0.f,0.f,0.f,0.f,0.f};
    v8f acc1 = acc0;

    #pragma unroll
    for (int ch = 0; ch < 8; ++ch) {
        FragB a, b0, b1;
        const unsigned short* ap = WvP + (((size_t)ct * 8 + ch) * 32 + lane) * 16;
        a.q[0] = *(const uint4*)(ap);
        a.q[1] = *(const uint4*)(ap + 8);
        const unsigned short* bp =
            XB + ((((size_t)b * 8 + ch) * 256 + kt2 * 2) * 32 + lane) * 16;
        b0.q[0] = *(const uint4*)(bp);
        b0.q[1] = *(const uint4*)(bp + 8);
        b1.q[0] = *(const uint4*)(bp + 512);
        b1.q[1] = *(const uint4*)(bp + 520);
        acc0 = __builtin_amdgcn_wmma_f32_16x16x32_bf16(false, a.v, false, b0.v,
                                                       (short)0, acc0, false, false);
        acc1 = __builtin_amdgcn_wmma_f32_16x16x32_bf16(false, a.v, false, b1.v,
                                                       (short)0, acc1, false, false);
    }

    // bias + stage tile in LDS as [c(16)][key(32)] bf16
    unsigned short* t = ldsT[wv];
    #pragma unroll
    for (int r = 0; r < 8; ++r) {
        float bb = bv[ct * 16 + r + 8 * lh];
        t[(r + 8 * lh) * 32 + lm]      = f2bf(acc0[r] + bb);
        t[(r + 8 * lh) * 32 + 16 + lm] = f2bf(acc1[r] + bb);
    }
    // re-read in V B-fragment order and store 32B/lane contiguous
    const unsigned short* src = t + lm * 32 + lh * 16;
    uint4 o0 = *(const uint4*)(src);
    uint4 o1 = *(const uint4*)(src + 8);
    unsigned short* dst = VB + ((((size_t)b * 128 + kt2) * 16 + ct) * 32 + lane) * 16;
    *(uint4*)(dst)     = o0;
    *(uint4*)(dst + 8) = o1;
}

// ---------------------------------------------------------------------------
// Flash attention: 1 wave per 16-query tile; 8 waves / block.
// Per 64-key iteration: 4 WMMAs for S, online softmax (DPP row reductions,
// row state in 8 regs: row = r + 8*(lane/16)), LDS re-layout of P (C-layout
// -> A-fragment), then 32 WMMAs for P * V^T into 16 f32 accumulators.
// ---------------------------------------------------------------------------
__global__ __launch_bounds__(256)
void attn_kernel(const unsigned short* __restrict__ Q,
                 const unsigned short* __restrict__ KB,
                 const unsigned short* __restrict__ VB,
                 float* __restrict__ O) {
    __shared__ unsigned short ldsP[8][16 * 64];   // per-wave 16x64 bf16 P staging

    const int lane = threadIdx.x & 31;
    const int wv   = threadIdx.x >> 5;
    const int b    = blockIdx.x >> 5;             // 32 query-blocks per batch
    const int qblk = blockIdx.x & 31;
    const int qbase = qblk * 128 + wv * 16;
    const int lm = lane & 15;
    const int lh = lane >> 4;

    FragB aq;
    {
        const unsigned short* qrow = Q + ((size_t)(b * Nn + qbase + lm)) * Dd;
        aq.q[0] = *(const uint4*)(qrow + lh * 16);       // k = lh*16 .. +7
        aq.q[1] = *(const uint4*)(qrow + lh * 16 + 8);   // k = lh*16+8 ..
    }

    v8f acc[16];
    #pragma unroll
    for (int ct = 0; ct < 16; ++ct)
        acc[ct] = (v8f){0.f,0.f,0.f,0.f,0.f,0.f,0.f,0.f};

    float mrow[8], lrow[8];
    #pragma unroll
    for (int r = 0; r < 8; ++r) { mrow[r] = -3.0e38f; lrow[r] = 0.f; }

    unsigned short* myP = ldsP[wv];
    const v8f zz = (v8f){0.f,0.f,0.f,0.f,0.f,0.f,0.f,0.f};

    for (int it = 0; it < Nn / 64; ++it) {
        // ---- S = Q * K^T for 64 keys ----
        const unsigned short* kp = KB + (((size_t)b * 256 + it * 4) * 32 + lane) * 16;
        v8f s[4];
        #pragma unroll
        for (int f = 0; f < 4; ++f) {
            FragB bk;
            bk.q[0] = *(const uint4*)(kp + f * 512);
            bk.q[1] = *(const uint4*)(kp + f * 512 + 8);
            s[f] = __builtin_amdgcn_wmma_f32_16x16x32_bf16(
                false, aq.v, false, bk.v, (short)0, zz, false, false);
        }
        __builtin_prefetch(kp + 4 * 512, 0, 1);   // next iter K frags

        // ---- online softmax (DPP row reductions, no LDS) ----
        float ef[8];
        #pragma unroll
        for (int r = 0; r < 8; ++r) {
            float tmx = fmaxf(fmaxf(s[0][r], s[1][r]), fmaxf(s[2][r], s[3][r]));
            tmx = rowmax16(tmx);
            float mn = fmaxf(mrow[r], tmx);
            ef[r] = __expf(mrow[r] - mn);
            mrow[r] = mn;
            lrow[r] *= ef[r];
        }
        #pragma unroll
        for (int ct = 0; ct < 16; ++ct)
            #pragma unroll
            for (int r = 0; r < 8; ++r)
                acc[ct][r] *= ef[r];

        #pragma unroll
        for (int r = 0; r < 8; ++r) {
            float rs = 0.f;
            #pragma unroll
            for (int f = 0; f < 4; ++f) {
                float p = __expf(s[f][r] - mrow[r]);
                rs += p;
                myP[(r + lh * 8) * 64 + f * 16 + lm] = f2bf(p);
            }
            lrow[r] += rowsum16(rs);
        }

        // ---- P back as two A-fragments (16 x 32 keys each) ----
        FragB ap[2];
        #pragma unroll
        for (int g = 0; g < 2; ++g) {
            const unsigned short* pr = myP + lm * 64 + g * 32 + lh * 16;
            ap[g].q[0] = *(const uint4*)(pr);
            ap[g].q[1] = *(const uint4*)(pr + 8);
        }

        // ---- O += P * V^T : 16 channel tiles x 2 key-groups ----
        const unsigned short* vp =
            VB + ((((size_t)b * 128 + it * 2) * 16) * 32 + lane) * 16;
        #pragma unroll
        for (int ct = 0; ct < 16; ++ct) {
            #pragma unroll
            for (int g = 0; g < 2; ++g) {
                const unsigned short* p = vp + ((size_t)g * 16 + ct) * 512;
                FragB bvf;
                bvf.q[0] = *(const uint4*)(p);
                bvf.q[1] = *(const uint4*)(p + 8);
                acc[ct] = __builtin_amdgcn_wmma_f32_16x16x32_bf16(
                    false, ap[g].v, false, bvf.v, (short)0, acc[ct], false, false);
            }
        }
        __builtin_prefetch(vp + (size_t)32 * 512, 0, 1);  // next iter V frags
    }

    float iv[8];
    #pragma unroll
    for (int r = 0; r < 8; ++r) iv[r] = 1.0f / lrow[r];
    #pragma unroll
    for (int ct = 0; ct < 16; ++ct)
        #pragma unroll
        for (int r = 0; r < 8; ++r) {
            int q = qbase + r + lh * 8;
            O[((size_t)(b * Nn + q)) * Cc + ct * 16 + lm] = acc[ct][r] * iv[r];
        }
}

// out = gamma * O * (mask>0.5) + x   (masked rows collapse to x)
__global__ void epilogue_kernel(const float* __restrict__ O, const float* __restrict__ x,
                                const float* __restrict__ mask, const float* __restrict__ gamma,
                                float* __restrict__ out) {
    size_t idx = (size_t)blockIdx.x * 256 + threadIdx.x;  // over B*N*C, c fastest
    int c = idx & (Cc - 1);
    size_t t = idx >> 8;
    int n = t & (Nn - 1);
    int b = t >> 12;
    float mb = mask[(size_t)b * Nn + n] > 0.5f ? 1.0f : 0.0f;
    size_t xi = ((size_t)b * Cc + c) * Nn + n;
    out[xi] = gamma[0] * O[idx] * mb + x[xi];
}

extern "C" void kernel_launch(void* const* d_in, const int* in_sizes, int n_in,
                              void* d_out, int out_size, void* d_ws, size_t ws_size,
                              hipStream_t stream) {
    const float* x     = (const float*)d_in[0];
    const float* mask  = (const float*)d_in[1];
    const float* Wq    = (const float*)d_in[2];
    const float* bq    = (const float*)d_in[3];
    const float* Wk    = (const float*)d_in[4];
    const float* bk    = (const float*)d_in[5];
    const float* Wv    = (const float*)d_in[6];
    const float* bv    = (const float*)d_in[7];
    const float* gamma = (const float*)d_in[8];
    float* out = (float*)d_out;

    // Workspace (bytes): Q 1MB | KB 1MB | VB 8MB | O f32 16MB (total ~27MB).
    // XB (8MB) + WvP (128KB) alias the O region: they are dead before attn writes O.
    char* ws = (char*)d_ws;
    unsigned short* Q   = (unsigned short*)(ws);
    unsigned short* KB  = (unsigned short*)(ws + ((size_t)1 << 20));
    unsigned short* VB  = (unsigned short*)(ws + ((size_t)2 << 20));
    float*          O   = (float*)(ws + ((size_t)10 << 20));
    unsigned short* XB  = (unsigned short*)(ws + ((size_t)10 << 20));
    unsigned short* WvP = (unsigned short*)(ws + ((size_t)18 << 20));

    proj_q_kernel<<<(Bb * Nn * Dd) / 256, 256, 0, stream>>>(x, Wq, bq, Q);
    proj_k_kernel<<<(Bb * (Nn / 16) * 32 * 16) / 256, 256, 0, stream>>>(x, Wk, bk, KB);
    pack_wv_kernel<<<(16 * 8 * 32 * 16) / 256, 256, 0, stream>>>(Wv, WvP);
    pack_x_kernel<<<(Bb * Cc * Nn) / 256, 256, 0, stream>>>(x, XB);
    proj_v_wmma_kernel<<<(Bb * 16 * 128) / 8, 256, 0, stream>>>(WvP, XB, bv, VB);
    attn_kernel<<<Bb * (Nn / 128), 256, 0, stream>>>(Q, KB, VB, O);
    epilogue_kernel<<<(Bb * Nn * Cc) / 256, 256, 0, stream>>>(O, x, mask, gamma, out);
}
